// TensorTrainODEBLOCK_19696720020138
// MI455X (gfx1250) — matrix operand order repaired and checked by reading, not compiled, not run on hardware
//
#include <hip/hip_runtime.h>

// ---------------------------------------------------------------------------
// Types for CDNA5 WMMA (wave32): bf16 16x16x32, f32 accumulate
// ---------------------------------------------------------------------------
typedef __attribute__((ext_vector_type(16))) __bf16 v16bf;
typedef __attribute__((ext_vector_type(8)))  float  v8f;

#define WMMA_BF16(A, B, C) \
  __builtin_amdgcn_wmma_f32_16x16x32_bf16(false, (A), false, (B), (short)0, (C), false, false)

// Problem dims (fixed by the reference)
#define BATCH 4096
#define DXDIM 128
#define DADIM 64
#define PDEG  8     // DEG+1
#define RRANK 32
#define HIDD  256
#define DOUTD 10
#define NSTEP 10

// ODE kernel geometry: 4 waves/block (keeps LDS <= 64KB, VGPRs <= 256)
#define ODE_WAVES 4

// ---------------------------------------------------------------------------
// A-operand build helpers.
// A k order chosen here (16x32, wave32; ISA table order differs): M=lane&15, K locals 16b..16b+15
// where b = lane>>4.  v16bf element e -> K=16b+e (matches packed B).
// So a lane's 16 A values are one run of 16 contiguous K -> four float4 loads.
// ---------------------------------------------------------------------------
static __device__ __forceinline__ void load16(float* va, const float* src, int b) {
  *(float4*)(va + 0)  = *(const float4*)(src + 16 * b);
  *(float4*)(va + 4)  = *(const float4*)(src + 16 * b + 4);
  *(float4*)(va + 8)  = *(const float4*)(src + 16 * b + 8);
  *(float4*)(va + 12) = *(const float4*)(src + 16 * b + 12);
}

static __device__ __forceinline__ v16bf make_a(const float* va, float scale) {
  v16bf a;
#pragma unroll
  for (int e = 0; e < 16; ++e) a[e] = (__bf16)(scale * va[e]);
  return a;
}

// ---------------------------------------------------------------------------
// Weight prepack kernels: f32 row-major -> bf16 packed in per-lane B-operand
// layout.  B k order chosen here (32x16; ISA table order differs): lane holds column N=lane&15, and K locals
// 16*(lane>>4) + (0..15) contiguous.  Packed block (kk,t) = 512 bf16:
// dst[((kk*NT + t)*32 + lane)*16 + e] = W[32*kk + 16*(lane>>4) + e][16*t + (lane&15)]
// -> inner-loop B fetch is one aligned 32-byte load per lane.
// ---------------------------------------------------------------------------
__global__ void pack_b_plain(const float* __restrict__ W, unsigned short* __restrict__ dst,
                             int Ntot, int NT, int total) {
  int id = blockIdx.x * blockDim.x + threadIdx.x;
  if (id >= total) return;
  int e = id & 15, lane = (id >> 4) & 31, blk = id >> 9;
  int t = blk % NT, kk = blk / NT;
  int b = lane >> 4, nl = lane & 15;
  int K = 32 * kk + 16 * b + e;
  int n = 16 * t + nl;
  float v = (n < Ntot) ? W[(size_t)K * Ntot + n] : 0.0f;
  ((__bf16*)dst)[id] = (__bf16)v;
}

// cores_mid: (63, R=32, P=8, R=32) -> 63 matrices coreMat[p*32+r][q]
__global__ void pack_mid(const float* __restrict__ src, unsigned short* __restrict__ dst,
                         int total) {
  int id = blockIdx.x * blockDim.x + threadIdx.x;
  if (id >= total) return;
  int e = id & 15, lane = (id >> 4) & 31, rem = id >> 9;  // rem = core*16 + kk*2 + t
  int t = rem & 1, kk = (rem >> 1) & 7, core = rem >> 4;
  int b = lane >> 4, nl = lane & 15;
  int r = 16 * b + e;          // K = 32*kk + 16b + e ; p = kk ; r = K & 31
  int p = kk;
  int q = 16 * t + nl;
  ((__bf16*)dst)[id] = (__bf16)src[(((size_t)core * 32 + r) * 8 + p) * 32 + q];
}

// core_last: (R=32, P=8, DA=64) -> coreLastMat[p*32+r][d], KB=8, NT=4
__global__ void pack_last(const float* __restrict__ src, unsigned short* __restrict__ dst,
                          int total) {
  int id = blockIdx.x * blockDim.x + threadIdx.x;
  if (id >= total) return;
  int e = id & 15, lane = (id >> 4) & 31, rem = id >> 9;  // rem = kk*4 + t
  int t = rem & 3, kk = rem >> 2;
  int b = lane >> 4, nl = lane & 15;
  int r = 16 * b + e, p = kk, d = 16 * t + nl;
  ((__bf16*)dst)[id] = (__bf16)src[((size_t)r * 8 + p) * 64 + d];
}

// ---------------------------------------------------------------------------
// One TT-ODE evaluation for a 16-row tile.
// zsW : per-wave LDS stage state, 16x64 f32 (plain row-major)
// vbW : per-wave LDS TT message,   16x32 f32
// kout: dz/dt in WMMA D layout, 4 N-tiles of v8f
// ---------------------------------------------------------------------------
static __device__ __forceinline__ void tt_eval(
    float t, const float* __restrict__ zsW, float* __restrict__ vbW,
    const float* __restrict__ core_first,
    const unsigned short* __restrict__ packMid,
    const unsigned short* __restrict__ packLast,
    int lane, int b, int ml, int nl, v8f kout[4]) {
  // v_init[r] = sum_p t^p * core_first[p][r]  (uniform over rows)
  {
    float tp = 1.f, s = 0.f;
#pragma unroll
    for (int p = 0; p < PDEG; ++p) { s += tp * core_first[p * RRANK + lane]; tp *= t; }
#pragma unroll
    for (int row = 0; row < 16; ++row) vbW[row * RRANK + lane] = s;
  }

  const v16bf* Bm = reinterpret_cast<const v16bf*>(packMid);

#pragma unroll 1
  for (int k = 0; k < DADIM - 1; ++k) {
    float va[16];
    load16(va, vbW + ml * RRANK, b);            // ds_load_b128 x4
    float z = zsW[ml * DADIM + k];
    float ph[PDEG];
    ph[0] = 1.f;
#pragma unroll
    for (int p = 1; p < PDEG; ++p) ph[p] = ph[p - 1] * z;

    v8f c0 = {}, c1 = {};
    const v16bf* bb = Bm + (size_t)k * 512 + lane;
    __builtin_prefetch(Bm + (size_t)(k + 1) * 512 + lane, 0, 1);  // global_prefetch_b8
    // unroll 4: at most 8 B-blocks (64 VGPRs) in flight -> stay under 256 VGPRs
#pragma unroll 4
    for (int kk = 0; kk < 8; ++kk) {            // K = 256 = 8 x 32; block kk <-> power p=kk
      v16bf a = make_a(va, ph[kk]);
      c0 = WMMA_BF16(a, bb[(2 * kk + 0) * 32], c0);
      c1 = WMMA_BF16(a, bb[(2 * kk + 1) * 32], c1);
    }
    // v_new back to LDS (D layout scatter: lane holds N=nl, rows vg+8b)
#pragma unroll
    for (int vg = 0; vg < 8; ++vg) {
      vbW[(vg + 8 * b) * RRANK + nl]      = c0[vg];
      vbW[(vg + 8 * b) * RRANK + 16 + nl] = c1[vg];
    }
  }

  // last core: (256 x 64), 4 N-tiles
  {
    float va[16];
    load16(va, vbW + ml * RRANK, b);
    float z = zsW[ml * DADIM + (DADIM - 1)];
    float ph[PDEG];
    ph[0] = 1.f;
#pragma unroll
    for (int p = 1; p < PDEG; ++p) ph[p] = ph[p - 1] * z;
    v8f zero = {};
#pragma unroll
    for (int tt = 0; tt < 4; ++tt) kout[tt] = zero;
    const v16bf* bl = reinterpret_cast<const v16bf*>(packLast) + lane;
    // unroll 2: 8 B-blocks in flight instead of 32 (accumulators already use 32 VGPRs)
#pragma unroll 2
    for (int kk = 0; kk < 8; ++kk) {
      v16bf a = make_a(va, ph[kk]);
#pragma unroll
      for (int tt = 0; tt < 4; ++tt)
        kout[tt] = WMMA_BF16(a, bl[(kk * 4 + tt) * 32], kout[tt]);
    }
  }
}

// ---------------------------------------------------------------------------
// Fused kernel: z0 = x@P, 10 RK4 steps of the TT ODE, write zf.
// 4 waves/block, 16 rows per wave.  RK4 state (zcur, acc) lives in per-wave
// LDS (not registers) so the 63-core WMMA scan runs with minimal VGPR
// pressure -> no spills, no extended-VGPR (s_set_vgpr_msb) mode.
// ---------------------------------------------------------------------------
__global__ void __launch_bounds__(128) tt_ode_kernel(
    const float* __restrict__ x, const float* __restrict__ core_first,
    const unsigned short* __restrict__ packP,
    const unsigned short* __restrict__ packMid,
    const unsigned short* __restrict__ packLast,
    float* __restrict__ zf) {
  __shared__ float zs_s[ODE_WAVES][16 * DADIM];   // stage input   16 KB
  __shared__ float zc_s[ODE_WAVES][16 * DADIM];   // current state 16 KB
  __shared__ float ac_s[ODE_WAVES][16 * DADIM];   // RK accumulator 16 KB
  __shared__ float vb_s[ODE_WAVES][16 * RRANK];   // TT message     8 KB
  const int lane = threadIdx.x & 31;
  const int wave = threadIdx.x >> 5;
  const int b = lane >> 4, ml = lane & 15, nl = ml;
  float* zsW = zs_s[wave];
  float* zcW = zc_s[wave];
  float* acW = ac_s[wave];
  float* vbW = vb_s[wave];
  const int gRow0 = blockIdx.x * (ODE_WAVES * 16) + wave * 16;

  // D-layout element address for (vg, tt) of this lane
#define ZIDX(vg, tt) ((vg + 8 * b) * DADIM + 16 * (tt) + nl)

  // ---- z0 = x @ P : K=128 (4 blocks), N=64 (4 tiles) ----
  {
    v8f zc[4];
    v8f zero = {};
#pragma unroll
    for (int tt = 0; tt < 4; ++tt) zc[tt] = zero;
    const float* xrow = x + (size_t)(gRow0 + ml) * DXDIM;
    const v16bf* Bp = reinterpret_cast<const v16bf*>(packP) + lane;
#pragma unroll 1
    for (int kb = 0; kb < 4; ++kb) {
      float va[16];
      load16(va, xrow + 32 * kb, b);
      v16bf a = make_a(va, 1.f);
#pragma unroll
      for (int tt = 0; tt < 4; ++tt) zc[tt] = WMMA_BF16(a, Bp[(kb * 4 + tt) * 32], zc[tt]);
    }
#pragma unroll
    for (int tt = 0; tt < 4; ++tt)
#pragma unroll
      for (int vg = 0; vg < 8; ++vg) {
        zcW[ZIDX(vg, tt)] = zc[tt][vg];
        zsW[ZIDX(vg, tt)] = zc[tt][vg];
      }
  }

  // ---- RK4 ----
  const float dt = 0.1f;
  float tcur = 0.f;
  for (int step = 0; step < NSTEP; ++step) {
#pragma unroll 1
    for (int s = 0; s < 4; ++s) {
      float ta  = (s == 0) ? 0.f : ((s == 3) ? 1.f : 0.5f);
      float wa  = (s == 1 || s == 2) ? (dt / 3.f) : (dt / 6.f);
      float wzc = (s == 2) ? dt : (dt * 0.5f);
      v8f kst[4];
      tt_eval(tcur + ta * dt, zsW, vbW, core_first, packMid, packLast, lane, b, ml, nl, kst);
      if (s == 0) {
#pragma unroll
        for (int tt = 0; tt < 4; ++tt)
#pragma unroll
          for (int vg = 0; vg < 8; ++vg) {
            float kv = kst[tt][vg];
            acW[ZIDX(vg, tt)] = wa * kv;
            zsW[ZIDX(vg, tt)] = zcW[ZIDX(vg, tt)] + wzc * kv;
          }
      } else if (s < 3) {
#pragma unroll
        for (int tt = 0; tt < 4; ++tt)
#pragma unroll
          for (int vg = 0; vg < 8; ++vg) {
            float kv = kst[tt][vg];
            acW[ZIDX(vg, tt)] += wa * kv;
            zsW[ZIDX(vg, tt)] = zcW[ZIDX(vg, tt)] + wzc * kv;
          }
      } else {
#pragma unroll
        for (int tt = 0; tt < 4; ++tt)
#pragma unroll
          for (int vg = 0; vg < 8; ++vg) {
            float kv = kst[tt][vg];
            float zn = zcW[ZIDX(vg, tt)] + acW[ZIDX(vg, tt)] + wa * kv;
            zcW[ZIDX(vg, tt)] = zn;
            zsW[ZIDX(vg, tt)] = zn;
          }
      }
    }
    tcur += dt;
  }

  // ---- write zf (plain row-major f32) ----
#pragma unroll
  for (int tt = 0; tt < 4; ++tt)
#pragma unroll
    for (int vg = 0; vg < 8; ++vg)
      zf[(size_t)(gRow0 + vg + 8 * b) * DADIM + 16 * tt + nl] = zcW[ZIDX(vg, tt)];
#undef ZIDX
}

// ---------------------------------------------------------------------------
// Generic WMMA GEMM + bias (+ReLU): out[B x Nout] = f(act[B x Ktot] @ W + bias)
// A operands built once per row tile, reused across all N tiles.
// ---------------------------------------------------------------------------
template <int KB, int NT, bool RELU>
__global__ void __launch_bounds__(256) mlp_kernel(
    const float* __restrict__ act, int Ktot,
    const unsigned short* __restrict__ packW,
    const float* __restrict__ bias,
    float* __restrict__ out, int Nout) {
  const int lane = threadIdx.x & 31;
  const int wave = threadIdx.x >> 5;
  const int b = lane >> 4, ml = lane & 15, nl = ml;
  const int gRow0 = blockIdx.x * 128 + wave * 16;

  const float* arow = act + (size_t)(gRow0 + ml) * Ktot;
  v16bf a[KB];
#pragma unroll
  for (int kb = 0; kb < KB; ++kb) {
    float va[16];
    load16(va, arow + 32 * kb, b);
    a[kb] = make_a(va, 1.f);
  }
  const v16bf* Bp = reinterpret_cast<const v16bf*>(packW) + lane;
#pragma unroll 2
  for (int t = 0; t < NT; ++t) {
    v8f c = {};
#pragma unroll
    for (int kb = 0; kb < KB; ++kb) c = WMMA_BF16(a[kb], Bp[(kb * NT + t) * 32], c);
    int col = 16 * t + nl;
    if (col < Nout) {
      float bv = bias[col];
#pragma unroll
      for (int vg = 0; vg < 8; ++vg) {
        float v = c[vg] + bv;
        if (RELU) v = fmaxf(v, 0.f);
        out[(size_t)(gRow0 + vg + 8 * b) * Nout + col] = v;
      }
    }
  }
}

// ---------------------------------------------------------------------------
// Host launcher
// ---------------------------------------------------------------------------
extern "C" void kernel_launch(void* const* d_in, const int* in_sizes, int n_in,
                              void* d_out, int out_size, void* d_ws, size_t ws_size,
                              hipStream_t stream) {
  (void)in_sizes; (void)n_in; (void)out_size; (void)ws_size;
  const float* x  = (const float*)d_in[0];
  const float* P  = (const float*)d_in[1];
  const float* cf = (const float*)d_in[2];
  const float* cm = (const float*)d_in[3];
  const float* cl = (const float*)d_in[4];
  const float* W1 = (const float*)d_in[5];
  const float* b1 = (const float*)d_in[6];
  const float* W2 = (const float*)d_in[7];
  const float* b2 = (const float*)d_in[8];
  const float* W3 = (const float*)d_in[9];
  const float* b3 = (const float*)d_in[10];
  float* out = (float*)d_out;

  char* ws = (char*)d_ws;
  size_t off = 0;
  auto take = [&](size_t bytes) {
    size_t o = off;
    off = (off + bytes + 255) & ~(size_t)255;
    return o;
  };
  unsigned short* packP  = (unsigned short*)(ws + take((size_t)4 * 4 * 512 * 2));   // P: KB=4, NT=4
  unsigned short* packM  = (unsigned short*)(ws + take((size_t)63 * 16 * 512 * 2)); // mid: 63 x KB=8,NT=2
  unsigned short* packL  = (unsigned short*)(ws + take((size_t)8 * 4 * 512 * 2));   // last: KB=8, NT=4
  unsigned short* packW1 = (unsigned short*)(ws + take((size_t)2 * 16 * 512 * 2));  // W1: KB=2, NT=16
  unsigned short* packW2 = (unsigned short*)(ws + take((size_t)8 * 16 * 512 * 2));  // W2: KB=8, NT=16
  unsigned short* packW3 = (unsigned short*)(ws + take((size_t)8 * 1 * 512 * 2));   // W3: KB=8, NT=1 (N padded)
  float* zf = (float*)(ws + take((size_t)BATCH * DADIM * 4));
  float* h1 = (float*)(ws + take((size_t)BATCH * HIDD * 4));
  float* h2 = (float*)(ws + take((size_t)BATCH * HIDD * 4));

  const int TPB = 256;
  // --- prepack weights to bf16 B-operand layout ---
  { int tot = 4 * 4 * 512;   pack_b_plain<<<(tot + TPB - 1) / TPB, TPB, 0, stream>>>(P,  packP,  DADIM, 4,  tot); }
  { int tot = 63 * 16 * 512; pack_mid    <<<(tot + TPB - 1) / TPB, TPB, 0, stream>>>(cm, packM,  tot); }
  { int tot = 8 * 4 * 512;   pack_last   <<<(tot + TPB - 1) / TPB, TPB, 0, stream>>>(cl, packL,  tot); }
  { int tot = 2 * 16 * 512;  pack_b_plain<<<(tot + TPB - 1) / TPB, TPB, 0, stream>>>(W1, packW1, HIDD, 16, tot); }
  { int tot = 8 * 16 * 512;  pack_b_plain<<<(tot + TPB - 1) / TPB, TPB, 0, stream>>>(W2, packW2, HIDD, 16, tot); }
  { int tot = 8 * 1 * 512;   pack_b_plain<<<(tot + TPB - 1) / TPB, TPB, 0, stream>>>(W3, packW3, DOUTD, 1, tot); }

  // --- fused z0 + RK4 TT-ODE : 4 waves/block, 16 rows/wave ---
  tt_ode_kernel<<<BATCH / (ODE_WAVES * 16), ODE_WAVES * 32, 0, stream>>>(
      x, cf, packP, packM, packL, zf);

  // --- terminal MLP ---
  mlp_kernel<2, 16, true ><<<BATCH / 128, TPB, 0, stream>>>(zf, DADIM, packW1, b1, h1, HIDD);
  mlp_kernel<8, 16, true ><<<BATCH / 128, TPB, 0, stream>>>(h1, HIDD,  packW2, b2, h2, HIDD);
  mlp_kernel<8, 1,  false><<<BATCH / 128, TPB, 0, stream>>>(h2, HIDD,  packW3, b3, out, DOUTD);
}
